// TransformerBlock_2568390443086
// MI455X (gfx1250) — matrix operand order repaired and checked by reading, not compiled, not run on hardware
//
#include <hip/hip_runtime.h>
#include <hip/hip_bf16.h>
#include <math.h>

// ---------------------------------------------------------------------------
// Transformer block (pre-LN, causal MHA + GELU FFN) for MI455X / gfx1250.
// All matmuls via V_WMMA_F32_16X16X32_F16 (wave32), flash-attention with
// in-register online softmax (v_permlane16 cross-lane reductions), f16
// activations/weights, f32 accumulation. K stored head-transposed ([B,H,HD,S]);
// WMMA B-fragments are gathered per element from LDS. Tile staging uses
// GLOBAL_LOAD_ASYNC_TO_LDS_B128 (ASYNCcnt) when the toolchain exposes it.
// ---------------------------------------------------------------------------

typedef _Float16 half_t;
typedef __attribute__((ext_vector_type(16))) _Float16 v16h;
typedef __attribute__((ext_vector_type(8)))  float    v8f;

#define B_   2
#define S_   2048
#define D_   1024
#define H_   16
#define HD_  64
#define R_   (B_ * S_)          // 4096 token rows

__device__ __forceinline__ v8f wmma_f16(v16h a, v16h b, v8f c) {
  return __builtin_amdgcn_wmma_f32_16x16x32_f16(false, a, false, b,
                                                (short)0, c, false, false);
}

union Frag16 { v16h v; half_t h[16]; uint4 u[2]; };
union Acc8   { v8f  v; float  f[8]; };
union H4     { half_t h[4]; uint2 u; };

__device__ __forceinline__ void zero_acc(Acc8& a) {
#pragma unroll
  for (int i = 0; i < 8; ++i) a.f[i] = 0.0f;
}

// ---------------------------------------------------------------------------
// Cross-lane reductions: v_permlane16_b32 (VALU) instead of ds_bpermute.
// Selectors encode lane^1 / lane^3 / lane^7 / lane^8 within a 16-lane row (acts as ^1/^2/^4/^8).
// ---------------------------------------------------------------------------
#if __has_builtin(__builtin_amdgcn_permlane16)
#define HAVE_PERM16 1
__device__ __forceinline__ float permx16(float v, unsigned lo, unsigned hi) {
  return __uint_as_float(__builtin_amdgcn_permlane16(
      __float_as_uint(v), __float_as_uint(v), lo, hi, false, false));
}
#else
#define HAVE_PERM16 0
#endif

__device__ __forceinline__ float red16_max(float v) {
#if HAVE_PERM16
  v = fmaxf(v, permx16(v, 0x67452301u, 0xEFCDAB89u));  // ^1
  v = fmaxf(v, permx16(v, 0x45670123u, 0xCDEF89ABu));  // ^3
  v = fmaxf(v, permx16(v, 0x01234567u, 0x89ABCDEFu));  // ^7
  v = fmaxf(v, permx16(v, 0xFEDCBA98u, 0x76543210u));  // ^8
#else
  v = fmaxf(v, __shfl_xor(v, 1));
  v = fmaxf(v, __shfl_xor(v, 2));
  v = fmaxf(v, __shfl_xor(v, 4));
  v = fmaxf(v, __shfl_xor(v, 8));
#endif
  return v;
}

__device__ __forceinline__ float red16_sum(float v) {
#if HAVE_PERM16
  v += permx16(v, 0x67452301u, 0xEFCDAB89u);
  v += permx16(v, 0x45670123u, 0xCDEF89ABu);
  v += permx16(v, 0x01234567u, 0x89ABCDEFu);
  v += permx16(v, 0xFEDCBA98u, 0x76543210u);
#else
  v += __shfl_xor(v, 1);
  v += __shfl_xor(v, 2);
  v += __shfl_xor(v, 4);
  v += __shfl_xor(v, 8);
#endif
  return v;
}

__device__ __forceinline__ float red32_sum(float v) {
  v = red16_sum(v);
#if __has_builtin(__builtin_amdgcn_permlanex16)
  v += __uint_as_float(__builtin_amdgcn_permlanex16(
      __float_as_uint(v), __float_as_uint(v), 0x76543210u, 0xFEDCBA98u,
      false, false));
#else
  v += __shfl_xor(v, 16);
#endif
  return v;
}

// ---------------------------------------------------------------------------
// Async global -> LDS staging (CDNA5 ASYNCcnt path) with safe fallback.
// Builtin signature (from hipcc diagnostic): param1 = int4-vector ptr in AS1
// (printed as "__device__"), param2 = AS3, then imm offset, imm cpol.
// ---------------------------------------------------------------------------
typedef __attribute__((__vector_size__(4 * sizeof(int)))) int vec4i_;

#if __has_builtin(__builtin_amdgcn_global_load_async_to_lds_b128) && \
    __has_builtin(__builtin_amdgcn_s_wait_asynccnt)
#define ASYNC_LDS 1
typedef __attribute__((address_space(1))) vec4i_ gvec4i;
typedef __attribute__((address_space(3))) vec4i_ lvec4i;
__device__ __forceinline__ void stage_b128(half_t* l, const half_t* g) {
  __builtin_amdgcn_global_load_async_to_lds_b128(
      (gvec4i*)(void*)const_cast<half_t*>(g),
      (lvec4i*)(void*)l, 0, 0);
}
__device__ __forceinline__ void stage_wait() {
  __builtin_amdgcn_s_wait_asynccnt(0);
}
#else
#define ASYNC_LDS 0
__device__ __forceinline__ void stage_b128(half_t* l, const half_t* g) {
  *(uint4*)l = *(const uint4*)g;
}
__device__ __forceinline__ void stage_wait() {}
#endif

// ---------------------------------------------------------------------------
// f32 -> f16 weight conversion (vectorized x4)
// ---------------------------------------------------------------------------
__global__ void cvt_pack3_kernel(const float* __restrict__ src,
                                 half_t* __restrict__ dst, int colOff) {
  int i4 = (blockIdx.x * blockDim.x + threadIdx.x) * 4;   // over D_*D_
  int r = i4 >> 10, c = i4 & (D_ - 1);
  const float4 v = *(const float4*)(src + i4);
  H4 o; o.h[0] = (half_t)v.x; o.h[1] = (half_t)v.y;
        o.h[2] = (half_t)v.z; o.h[3] = (half_t)v.w;
  *(uint2*)(dst + (size_t)r * (3 * D_) + colOff + c) = o.u;
}

__global__ void cvt_plain_kernel(const float* __restrict__ src,
                                 half_t* __restrict__ dst) {
  int i4 = (blockIdx.x * blockDim.x + threadIdx.x) * 4;
  const float4 v = *(const float4*)(src + i4);
  H4 o; o.h[0] = (half_t)v.x; o.h[1] = (half_t)v.y;
        o.h[2] = (half_t)v.z; o.h[3] = (half_t)v.w;
  *(uint2*)(dst + i4) = o.u;
}

// ---------------------------------------------------------------------------
// LayerNorm (one 256-thread block per row of 1024) -> f16 output
// ---------------------------------------------------------------------------
__global__ __launch_bounds__(256)
void layernorm_f16_kernel(const float* __restrict__ x,
                          const float* __restrict__ g,
                          const float* __restrict__ be,
                          half_t* __restrict__ out) {
  __shared__ float red[16];
  const int row = blockIdx.x;
  const int tid = threadIdx.x;
  const float4 xv = ((const float4*)(x + (size_t)row * D_))[tid];
  float s  = red32_sum(xv.x + xv.y + xv.z + xv.w);
  float s2 = red32_sum(xv.x * xv.x + xv.y * xv.y + xv.z * xv.z + xv.w * xv.w);
  if ((tid & 31) == 0) { red[tid >> 5] = s; red[8 + (tid >> 5)] = s2; }
  __syncthreads();
  if (tid == 0) {
    float S = 0.f, S2 = 0.f;
#pragma unroll
    for (int i = 0; i < 8; ++i) { S += red[i]; S2 += red[8 + i]; }
    float mu  = S * (1.0f / D_);
    float var = S2 * (1.0f / D_) - mu * mu;
    red[0] = mu;
    red[8] = rsqrtf(var + 1e-5f);
  }
  __syncthreads();
  const float mu = red[0], rs = red[8];
  const float4 gv = ((const float4*)g)[tid];
  const float4 bv = ((const float4*)be)[tid];
  H4 o;
  o.h[0] = (half_t)((xv.x - mu) * rs * gv.x + bv.x);
  o.h[1] = (half_t)((xv.y - mu) * rs * gv.y + bv.y);
  o.h[2] = (half_t)((xv.z - mu) * rs * gv.z + bv.z);
  o.h[3] = (half_t)((xv.w - mu) * rs * gv.w + bv.w);
  *(uint2*)(out + (size_t)row * D_ + tid * 4) = o.u;
}

// ---------------------------------------------------------------------------
// WMMA GEMM: C[M,N] = A[M,K]*B[K,N], f16 in / f32 accum.
// Block tile 128x128, K-tile 32, 256 threads = 8 waves (4 Mwaves x 2 Nwaves),
// each wave owns a 32x64 tile = 2x4 WMMA accumulators.
// Epilogues: 0 = QKV scatter (q scaled 1/sqrt(HD); k stored transposed),
//            1 = bias+residual->f32, 2 = bias+GELU->f16.
// ---------------------------------------------------------------------------
#define LDA_ 40     // 32 + 8 halves pad
#define LDB_ 136    // 128 + 8 halves pad

template <int EPI>
__global__ __launch_bounds__(256)
void gemm_wmma_kernel(const half_t* __restrict__ A, const half_t* __restrict__ Bm,
                      int M, int N, int K,
                      const float* __restrict__ bias, const float* __restrict__ res,
                      float* __restrict__ outF, half_t* __restrict__ outH,
                      half_t* __restrict__ q_out, half_t* __restrict__ k_out,
                      half_t* __restrict__ v_out) {
  __shared__ alignas(16) half_t sA[128 * LDA_];
  __shared__ alignas(16) half_t sB[32 * LDB_];

  const int tid  = threadIdx.x;
  const int lane = tid & 31;
  const int wave = tid >> 5;
  const int mw   = wave & 3;           // 0..3
  const int nw   = wave >> 2;          // 0..1
  const int m16  = lane & 15;
  const int hi   = lane >> 4;
  const int kh8  = hi * 8;
  const int bm   = blockIdx.y * 128;
  const int bn   = blockIdx.x * 128;

  Acc8 acc[2][4];
#pragma unroll
  for (int mi = 0; mi < 2; ++mi)
#pragma unroll
    for (int ni = 0; ni < 4; ++ni) zero_acc(acc[mi][ni]);

  const int ra = tid >> 2, ca = (tid & 3) * 8;     // A tile: 128 x 32
  const int rb = tid >> 4, cb = (tid & 15) * 8;    // B tile: 32 x 128

  for (int kt = 0; kt < K; kt += 32) {
    stage_b128(&sA[ra * LDA_ + ca],        A + (size_t)(bm + ra) * K + kt + ca);
    stage_b128(&sA[(ra + 64) * LDA_ + ca], A + (size_t)(bm + ra + 64) * K + kt + ca);
    stage_b128(&sB[rb * LDB_ + cb],        Bm + (size_t)(kt + rb) * N + bn + cb);
    stage_b128(&sB[(rb + 16) * LDB_ + cb], Bm + (size_t)(kt + rb + 16) * N + bn + cb);
    if (kt + 32 < K) {
      __builtin_prefetch(A + (size_t)(bm + ra) * K + kt + 32 + ca, 0, 3);
      __builtin_prefetch(Bm + (size_t)(kt + 32 + rb) * N + bn + cb, 0, 3);
    }
    stage_wait();
    __syncthreads();

    Frag16 fa[2], fb[4];
#pragma unroll
    for (int mi = 0; mi < 2; ++mi) {
      const half_t* ap = &sA[(mw * 32 + mi * 16 + m16) * LDA_];
      fa[mi].u[0] = *(const uint4*)(ap + kh8);        // K = kh8 .. kh8+7
      fa[mi].u[1] = *(const uint4*)(ap + 16 + kh8);   // K = 16+kh8 ..
    }
#pragma unroll
    for (int ni = 0; ni < 4; ++ni) {
      const half_t* bp = &sB[nw * 64 + ni * 16 + m16]; // col N = m16, K = kh8.., 16+kh8..
#pragma unroll
      for (int s = 0; s < 8; ++s) {
        fb[ni].h[s]     = bp[(kh8 + s) * LDB_];
        fb[ni].h[8 + s] = bp[(16 + kh8 + s) * LDB_];
      }
    }
#pragma unroll
    for (int mi = 0; mi < 2; ++mi)
#pragma unroll
      for (int ni = 0; ni < 4; ++ni)
        acc[mi][ni].v = wmma_f16(fa[mi].v, fb[ni].v, acc[mi][ni].v);
    __syncthreads();
  }

  // Epilogue: C element (r, lane) -> row = r + 8*hi, col = m16 within tile.
#pragma unroll
  for (int mi = 0; mi < 2; ++mi) {
#pragma unroll
    for (int ni = 0; ni < 4; ++ni) {
#pragma unroll
      for (int r = 0; r < 8; ++r) {
        const int row = bm + mw * 32 + mi * 16 + r + 8 * hi;
        const int col = bn + nw * 64 + ni * 16 + m16;
        float v = acc[mi][ni].f[r];
        if constexpr (EPI == 0) {
          const int which = col >> 10;          // 0=q 1=k 2=v
          const int d     = col & (D_ - 1);
          const int head  = d >> 6;
          const int hd    = d & 63;
          const int b     = row >> 11;
          const int s     = row & (S_ - 1);
          if (which == 0) {
            q_out[(((size_t)(b * H_ + head)) * S_ + s) * HD_ + hd] =
                (half_t)(v * 0.125f);           // 1/sqrt(HD) softmax scale
          } else if (which == 1) {
            // K stored transposed per head: [B,H,HD,S]
            k_out[(((size_t)(b * H_ + head)) * HD_ + hd) * S_ + s] = (half_t)v;
          } else {
            v_out[(((size_t)(b * H_ + head)) * S_ + s) * HD_ + hd] = (half_t)v;
          }
        } else if constexpr (EPI == 1) {
          outF[(size_t)row * N + col] =
              v + bias[col] + res[(size_t)row * N + col];
        } else {  // EPI == 2 : bias + exact GELU -> f16
          float t = v + bias[col];
          float ge = 0.5f * t * (1.0f + erff(t * 0.70710678118654752f));
          outH[(size_t)row * N + col] = (half_t)ge;
        }
      }
    }
  }
}

// ---------------------------------------------------------------------------
// Flash attention (causal), all-WMMA. Block = 128 threads (4 waves),
// block handles 64 q rows of one (b,h); wave owns 16 q rows.
// Q/V are [B,H,S,64] f16 (q pre-scaled by 1/sqrt(HD)); K is [B,H,64,S] f16
// (head-transposed); score B-fragments are strided LDS gathers. ctx -> [B,S,D] f16.
// ---------------------------------------------------------------------------
#define LDKT_ 40    // kT tile row: 32 kpos + 8 pad
#define LDV_  72    // v tile row: 64 hd + 8 pad
#define LDP_  40    // p tile row: 32 + 8 pad

__global__ __launch_bounds__(128)
void attn_flash_kernel(const half_t* __restrict__ Q, const half_t* __restrict__ KT,
                       const half_t* __restrict__ V, half_t* __restrict__ ctx) {
  __shared__ alignas(16) half_t sKT[64 * LDKT_];   // [hd][kpos]
  __shared__ alignas(16) half_t sV[32 * LDV_];     // [kpos][hd]
  __shared__ alignas(16) half_t sP[4 * 16 * LDP_]; // per-wave P tiles

  const int tid  = threadIdx.x;
  const int lane = tid & 31;
  const int wave = tid >> 5;
  const int m16  = lane & 15;
  const int hi   = lane >> 4;
  const int kh8  = hi * 8;
  const int qb   = blockIdx.x * 64;
  const int h    = blockIdx.y;
  const int b    = blockIdx.z;
  const size_t headQV = ((size_t)(b * H_ + h)) * S_ * HD_;
  const size_t headKT = ((size_t)(b * H_ + h)) * HD_ * S_;
  const int qrow = qb + wave * 16;

  // q A-fragments (K=0..31 and K=32..63 of head_dim) -- contiguous 16B loads
  Frag16 fq0, fq1;
  {
    const half_t* qr = Q + headQV + (size_t)(qrow + m16) * HD_;
    fq0.u[0] = *(const uint4*)(qr + kh8);
    fq0.u[1] = *(const uint4*)(qr + 16 + kh8);
    fq1.u[0] = *(const uint4*)(qr + 32 + kh8);
    fq1.u[1] = *(const uint4*)(qr + 48 + kh8);
  }

  Acc8 O[4];
#pragma unroll
  for (int n4 = 0; n4 < 4; ++n4) zero_acc(O[n4]);
  float mrow[8], lrow[8];
#pragma unroll
  for (int r = 0; r < 8; ++r) { mrow[r] = -1e30f; lrow[r] = 0.0f; }

  half_t* pw = &sP[wave * 16 * LDP_];
  const int kend = qb + 64;
  const int rk = tid >> 2, ck = (tid & 3) * 8;   // kT tile: 64 rows x 32
  const int rv = tid >> 3, cv = (tid & 7) * 8;   // v  tile: 32 rows x 64

  for (int kt = 0; kt < kend; kt += 32) {
    // stage K^T [64 hd][32 kpos] and V [32 kpos][64 hd] tiles into LDS
    stage_b128(&sKT[rk * LDKT_ + ck],
               KT + headKT + (size_t)rk * S_ + kt + ck);
    stage_b128(&sKT[(rk + 32) * LDKT_ + ck],
               KT + headKT + (size_t)(rk + 32) * S_ + kt + ck);
    stage_b128(&sV[rv * LDV_ + cv],
               V + headQV + (size_t)(kt + rv) * HD_ + cv);
    stage_b128(&sV[(rv + 16) * LDV_ + cv],
               V + headQV + (size_t)(kt + rv + 16) * HD_ + cv);
    stage_wait();
    __syncthreads();

    // scores s[n] = q . k^T for key columns [kt+16n, kt+16n+16)
    Acc8 sc[2];
#pragma unroll
    for (int n = 0; n < 2; ++n) {
      zero_acc(sc[n]);
      Frag16 fk0, fk1;  // B = k^T[hd][kpos]; frag col = kpos = m16, elems over hd
      const half_t* b0 = &sKT[n * 16 + m16];                  // hd 0..31
      const half_t* b1 = &sKT[32 * LDKT_ + n * 16 + m16];     // hd 32..63
#pragma unroll
      for (int s = 0; s < 8; ++s) {
        fk0.h[s]     = b0[(kh8 + s) * LDKT_];
        fk0.h[8 + s] = b0[(16 + kh8 + s) * LDKT_];
        fk1.h[s]     = b1[(kh8 + s) * LDKT_];
        fk1.h[8 + s] = b1[(16 + kh8 + s) * LDKT_];
      }
      sc[n].v = wmma_f16(fq0.v, fk0.v, sc[n].v);
      sc[n].v = wmma_f16(fq1.v, fk1.v, sc[n].v);
    }

    // online softmax update (rows r+8*hi, cols m16 / 16+m16); all cross-lane
    // reductions are v_permlane16 (VALU), no LDS traffic.
#pragma unroll
    for (int r = 0; r < 8; ++r) {
      const int qa = qrow + r + 8 * hi;
      const int c0 = kt + m16, c1 = kt + 16 + m16;
      float s0 = (c0 <= qa) ? sc[0].f[r] : -1e30f;
      float s1 = (c1 <= qa) ? sc[1].f[r] : -1e30f;
      const float mx = red16_max(fmaxf(s0, s1));
      const float mn = fmaxf(mrow[r], mx);
      const float p0 = __expf(s0 - mn);
      const float p1 = __expf(s1 - mn);
      const float rs = red16_sum(p0 + p1);
      const float scale = __expf(mrow[r] - mn);
      lrow[r] = lrow[r] * scale + rs;
      mrow[r] = mn;
#pragma unroll
      for (int n4 = 0; n4 < 4; ++n4) O[n4].f[r] *= scale;
      const int prow = r + 8 * hi;
      pw[prow * LDP_ + m16]      = (half_t)p0;
      pw[prow * LDP_ + 16 + m16] = (half_t)p1;
    }
    // same-wave LDS RAW: DS ops are in-order per wave; fence the compiler and
    // drain DScnt before re-reading P as an A-fragment.
    asm volatile("s_wait_dscnt 0" ::: "memory");

    Frag16 fp;
    {
      const half_t* pr = &pw[m16 * LDP_];
      fp.u[0] = *(const uint4*)(pr + kh8);
      fp.u[1] = *(const uint4*)(pr + 16 + kh8);
    }
#pragma unroll
    for (int n4 = 0; n4 < 4; ++n4) {
      Frag16 fv;  // B = v[kpos][hd]; frag col = hd = m16, elems over kpos
      const half_t* vr = &sV[n4 * 16 + m16];
#pragma unroll
      for (int s = 0; s < 8; ++s) {
        fv.h[s]     = vr[(kh8 + s) * LDV_];
        fv.h[8 + s] = vr[(16 + kh8 + s) * LDV_];
      }
      O[n4].v = wmma_f16(fp.v, fv.v, O[n4].v);
    }
    __syncthreads();
  }

  // normalize + write ctx as [B,S,D] f16
#pragma unroll
  for (int n4 = 0; n4 < 4; ++n4) {
#pragma unroll
    for (int r = 0; r < 8; ++r) {
      const int srow = qrow + r + 8 * hi;
      const int col  = h * HD_ + n4 * 16 + m16;
      ctx[((size_t)b * S_ + srow) * D_ + col] =
          (half_t)(O[n4].f[r] * (1.0f / lrow[r]));
    }
  }
}

// ---------------------------------------------------------------------------
// Host-side orchestration
// ---------------------------------------------------------------------------
extern "C" void kernel_launch(void* const* d_in, const int* in_sizes, int n_in,
                              void* d_out, int out_size, void* d_ws, size_t ws_size,
                              hipStream_t stream) {
  (void)in_sizes; (void)n_in; (void)out_size; (void)ws_size;
  const float* x   = (const float*)d_in[0];
  const float* Wq  = (const float*)d_in[1];
  const float* Wk  = (const float*)d_in[2];
  const float* Wv  = (const float*)d_in[3];
  const float* Wo  = (const float*)d_in[4];
  const float* bo  = (const float*)d_in[5];
  const float* g1  = (const float*)d_in[6];
  const float* be1 = (const float*)d_in[7];
  const float* g2  = (const float*)d_in[8];
  const float* be2 = (const float*)d_in[9];
  const float* W1  = (const float*)d_in[10];
  const float* bf1 = (const float*)d_in[11];
  const float* W2  = (const float*)d_in[12];
  const float* bf2 = (const float*)d_in[13];
  float* out = (float*)d_out;

  char* ws = (char*)d_ws;
  size_t off = 0;
  auto take = [&](size_t bytes) {
    void* p = ws + off;
    off += (bytes + 255) & ~(size_t)255;
    return p;
  };
  half_t* hF    = (half_t*)take((size_t)R_ * D_ * 2);          //  8 MB
  half_t* Wqkv  = (half_t*)take((size_t)D_ * 3 * D_ * 2);      //  6 MB
  half_t* qF    = (half_t*)take((size_t)R_ * D_ * 2);          //  8 MB
  half_t* kF    = (half_t*)take((size_t)R_ * D_ * 2);          //  8 MB (transposed)
  half_t* vF    = (half_t*)take((size_t)R_ * D_ * 2);          //  8 MB
  half_t* ctxF  = (half_t*)take((size_t)R_ * D_ * 2);          //  8 MB
  half_t* WoF   = (half_t*)take((size_t)D_ * D_ * 2);          //  2 MB
  float*  x1    = (float*)take((size_t)R_ * D_ * 4);           // 17 MB
  half_t* h2F   = (half_t*)take((size_t)R_ * D_ * 2);          //  8 MB
  half_t* W1F   = (half_t*)take((size_t)D_ * 4 * D_ * 2);      //  8 MB
  half_t* ff1F  = (half_t*)take((size_t)R_ * 4 * D_ * 2);      // 34 MB
  half_t* W2F   = (half_t*)take((size_t)4 * D_ * D_ * 2);      //  8 MB

  // weight conversion (4 elements per thread)
  cvt_pack3_kernel<<<(D_ * D_) / 1024, 256, 0, stream>>>(Wq, Wqkv, 0);
  cvt_pack3_kernel<<<(D_ * D_) / 1024, 256, 0, stream>>>(Wk, Wqkv, D_);
  cvt_pack3_kernel<<<(D_ * D_) / 1024, 256, 0, stream>>>(Wv, Wqkv, 2 * D_);
  cvt_plain_kernel<<<(D_ * D_) / 1024, 256, 0, stream>>>(Wo, WoF);
  cvt_plain_kernel<<<(D_ * 4 * D_) / 1024, 256, 0, stream>>>(W1, W1F);
  cvt_plain_kernel<<<(4 * D_ * D_) / 1024, 256, 0, stream>>>(W2, W2F);

  // LN1 -> h (f16)
  layernorm_f16_kernel<<<R_, 256, 0, stream>>>(x, g1, be1, hF);

  // QKV projection: [4096x1024] x [1024x3072], scatter to q/kT/v heads
  {
    dim3 grid(3 * D_ / 128, R_ / 128);
    gemm_wmma_kernel<0><<<grid, 256, 0, stream>>>(
        hF, Wqkv, R_, 3 * D_, D_, nullptr, nullptr, nullptr, nullptr,
        qF, kF, vF);
  }

  // causal flash attention
  {
    dim3 grid(S_ / 64, H_, B_);
    attn_flash_kernel<<<grid, 128, 0, stream>>>(qF, kF, vF, ctxF);
  }

  // out projection + bias + residual -> x1 (f32)
  {
    dim3 grid(D_ / 128, R_ / 128);
    gemm_wmma_kernel<1><<<grid, 256, 0, stream>>>(
        ctxF, WoF, R_, D_, D_, bo, x, x1, nullptr, nullptr, nullptr, nullptr);
  }

  // LN2 -> h2 (f16)
  layernorm_f16_kernel<<<R_, 256, 0, stream>>>(x1, g2, be2, h2F);

  // FFN1: bias + exact GELU -> f16
  {
    dim3 grid(4 * D_ / 128, R_ / 128);
    gemm_wmma_kernel<2><<<grid, 256, 0, stream>>>(
        h2F, W1F, R_, 4 * D_, D_, bf1, nullptr, nullptr, ff1F,
        nullptr, nullptr, nullptr);
  }

  // FFN2: bias + residual(x1) -> final f32 output
  {
    dim3 grid(D_ / 128, R_ / 128);
    gemm_wmma_kernel<1><<<grid, 256, 0, stream>>>(
        ff1F, W2F, R_, D_, 4 * D_, bf2, x1, out, nullptr,
        nullptr, nullptr, nullptr);
  }
}
